// GAT_25451976196185
// MI455X (gfx1250) — matrix operand order repaired and checked by reading, not compiled, not run on hardware
//
#include <hip/hip_runtime.h>

// ---------------------------------------------------------------------------
// GAT forward (5 stacked GATConv layers) for MI455X / gfx1250.
// GEMMs: double-buffered async-LDS staged bf16 WMMA (v_wmma_f32_16x16x32_bf16),
// W pre-transposed so both A and B fragments are 2x ds_load_b128.
// Edge softmax/aggregation: global_atomic_max_i32 / global_atomic_add_f32.
// ---------------------------------------------------------------------------

#define NODES 20000
#define EDGES 160000
#define NEG_SLOPE 0.2f

typedef __attribute__((ext_vector_type(16))) __bf16 v16bf;
typedef __attribute__((ext_vector_type(8)))  float  v8f;

#if defined(__gfx1250__) && __has_builtin(__builtin_amdgcn_global_load_async_to_lds_b128)
#define USE_ASYNC_LDS 1
#else
#define USE_ASYNC_LDS 0
#endif

// stage one 16-byte chunk from global memory into LDS (async DMA when present)
__device__ __forceinline__ void stage16(const void* gptr, void* lptr) {
#if USE_ASYNC_LDS
    typedef int v4i_g __attribute__((vector_size(16)));
    typedef __attribute__((address_space(1))) v4i_g gas_v4i;
    typedef __attribute__((address_space(3))) v4i_g las_v4i;
    // generic shared pointer: addr[31:0] IS the LDS byte offset
    __builtin_amdgcn_global_load_async_to_lds_b128(
        (gas_v4i*)(uintptr_t)gptr,
        (las_v4i*)(uintptr_t)(unsigned)(uintptr_t)lptr,
        /*offset=*/0, /*cpol=*/0);
#else
    *(float4*)lptr = *(const float4*)gptr;
#endif
}

__device__ __forceinline__ void stage_wait() {
#if USE_ASYNC_LDS
#if __has_builtin(__builtin_amdgcn_s_wait_asynccnt)
    __builtin_amdgcn_s_wait_asynccnt(0);
#else
    asm volatile("s_wait_asynccnt 0x0" ::: "memory");
#endif
#endif
    __syncthreads();
}

// ---------------- f32 -> bf16 convert (row-major copy) ---------------------
__global__ __launch_bounds__(256) void k_f32_to_bf16(const float* __restrict__ in,
                                                     __bf16* __restrict__ out,
                                                     size_t n) {
    size_t i = (size_t)blockIdx.x * blockDim.x + threadIdx.x;
    if (i < n) out[i] = (__bf16)in[i];
}

// ---------------- f32 [K,F] -> bf16 transposed [F,K] -----------------------
__global__ __launch_bounds__(256) void k_f32_to_bf16_T(const float* __restrict__ in,
                                                       __bf16* __restrict__ out,
                                                       int K, int F) {
    int idx = blockIdx.x * blockDim.x + threadIdx.x;
    if (idx >= K * F) return;
    int k = idx / F, f = idx - k * F;               // coalesced read over f
    out[(size_t)f * K + k] = (__bf16)in[idx];
}

// ---------------- WMMA GEMM: C[M,F] = A[M,K] * Wt[F,K]^T -------------------
// Block: 256 threads (8 waves), tile BM=32 x BN, K-step 32, double-buffered.
// Waves: 2 along M x 4 along N; each wave computes 16 x (TN*16) with TN WMMAs
// per K-step sharing one A fragment. Requires M%32==0, K%32==0, F%BN==0.
template <int BN, int TN>
__global__ __launch_bounds__(256) void k_gemm_bf16_wmma(const __bf16* __restrict__ A,
                                                        const __bf16* __restrict__ Wt,
                                                        float* __restrict__ C,
                                                        int M, int K, int F) {
    // padded rows (40 elems = 80 B): 16B-aligned chunks, bank-spread
    __shared__ __bf16 As[2][32][40];
    __shared__ __bf16 Bt[2][BN][40];

    const int tid    = threadIdx.x;
    const int lane   = tid & 31;
    const int waveId = tid >> 5;       // 0..7
    const int waveM  = waveId & 1;     // 2 waves along M
    const int waveN  = waveId >> 1;    // 4 waves along N
    const int half   = lane >> 4;
    const int l16    = lane & 15;
    const int m0     = blockIdx.x * 32;
    const int n0     = blockIdx.y * BN;

    auto stage = [&](int buf, int kk) {
        // A tile 32x32: 128 x 16B chunks (threads 0..127)
        if (tid < 128) {
            const int arow = tid >> 2;
            const int acol = (tid & 3) * 8;
            stage16(A + (size_t)(m0 + arow) * K + kk + acol, &As[buf][arow][acol]);
        }
        // B tile (K-contiguous per output column): BN x 32 -> BN*4 chunks
        constexpr int BCH = BN * 4;
#pragma unroll
        for (int p = 0; p < BCH / 256; ++p) {
            const int ci   = tid + p * 256;
            const int brow = ci >> 2;
            const int bcol = (ci & 3) * 8;
            stage16(Wt + (size_t)(n0 + brow) * K + kk + bcol, &Bt[buf][brow][bcol]);
        }
    };

    v8f c[TN] = {};
    const int nIter = K >> 5;
    int ibuf = 0;
    stage(0, 0);
    for (int it = 0; it < nIter; ++it) {
        stage_wait();                                   // drain async + barrier
        if (it + 1 < nIter) stage(ibuf ^ 1, (it + 1) << 5);  // overlap next tile

        union { v16bf v; float4 f[2]; } ua, ub;
        const int am = waveM * 16 + l16;
        ua.f[0] = *(const float4*)&As[ibuf][am][half * 8];
        ua.f[1] = *(const float4*)&As[ibuf][am][16 + half * 8];
#pragma unroll
        for (int t = 0; t < TN; ++t) {
            const int n = waveN * TN * 16 + t * 16 + l16;
            ub.f[0] = *(const float4*)&Bt[ibuf][n][half * 8];
            ub.f[1] = *(const float4*)&Bt[ibuf][n][16 + half * 8];
            c[t] = __builtin_amdgcn_wmma_f32_16x16x32_bf16(
                       false, ua.v, false, ub.v, (short)0, c[t], false, false);
        }
        ibuf ^= 1;
    }

    // store: C VGPR r -> row = r + 8*half, col striped over lanes
#pragma unroll
    for (int t = 0; t < TN; ++t) {
        const int n = n0 + waveN * TN * 16 + t * 16 + l16;
#pragma unroll
        for (int r = 0; r < 8; ++r)
            C[(size_t)(m0 + waveM * 16 + r + 8 * half) * F + n] = c[t][r];
    }
}

// ---------------- attention logits: al_s/al_d[n,h] = <h[n,h,:], a[h,:]> ----
__global__ __launch_bounds__(256) void k_attn_logits(const float* __restrict__ hf,
                                                     const float* __restrict__ a_src,
                                                     const float* __restrict__ a_dst,
                                                     float* __restrict__ als,
                                                     float* __restrict__ ald,
                                                     int Nn, int Hh, int Cc) {
    int idx = blockIdx.x * blockDim.x + threadIdx.x;
    if (idx >= Nn * Hh) return;
    int n = idx / Hh, hh = idx - n * Hh;
    const float* hp = hf + ((size_t)n * Hh + hh) * Cc;
    const float* as = a_src + (size_t)hh * Cc;
    const float* ad = a_dst + (size_t)hh * Cc;
    float s = 0.f, d = 0.f;
    for (int c = 0; c < Cc; ++c) { float v = hp[c]; s = fmaf(v, as[c], s); d = fmaf(v, ad[c], d); }
    als[idx] = s;
    ald[idx] = d;
}

// ---------------- init: mkey = INT_MIN, denom = 0 --------------------------
__global__ __launch_bounds__(256) void k_init_attn(int* __restrict__ mkey,
                                                   float* __restrict__ denom, int n) {
    int i = blockIdx.x * blockDim.x + threadIdx.x;
    if (i < n) { mkey[i] = (int)0x80000000; denom[i] = 0.f; }
}

__global__ __launch_bounds__(256) void k_zero_f32(float* __restrict__ p, size_t n) {
    size_t i = (size_t)blockIdx.x * blockDim.x + threadIdx.x;
    if (i < n) p[i] = 0.f;
}

__global__ __launch_bounds__(256) void k_copy_f32(float* __restrict__ dst,
                                                  const float* __restrict__ src, size_t n) {
    size_t i = (size_t)blockIdx.x * blockDim.x + threadIdx.x;
    if (i < n) dst[i] = src[i];
}

// monotone float<->int ordering keys (for integer atomicMax on floats)
__device__ __forceinline__ int f2key(float f) {
    int b = __float_as_int(f);
    return b >= 0 ? b : (b ^ 0x7fffffff);
}
__device__ __forceinline__ float key2f(int k) {
    return __int_as_float(k >= 0 ? k : (k ^ 0x7fffffff));
}

// self-loops appended after the E real edges: e >= E  ->  src = dst = e - E
__device__ __forceinline__ void edge_sd(const int* __restrict__ ei, int e, int& s, int& d) {
    if (e < EDGES) { s = ei[e]; d = ei[EDGES + e]; } else { s = d = e - EDGES; }
}

// ---- pass 1: e = leaky_relu(als[src]+ald[dst]); segment max over dst ------
__global__ __launch_bounds__(256) void k_edge_max(const int* __restrict__ ei,
                                                  const float* __restrict__ als,
                                                  const float* __restrict__ ald,
                                                  float* __restrict__ ev,
                                                  int* __restrict__ mkey, int Hh) {
    int idx = blockIdx.x * blockDim.x + threadIdx.x;
    int tot = (EDGES + NODES) * Hh;
    if (idx >= tot) return;
    int e = idx / Hh, hh = idx - e * Hh;
    int s, d; edge_sd(ei, e, s, d);
    float v = als[s * Hh + hh] + ald[d * Hh + hh];
    v = v > 0.f ? v : NEG_SLOPE * v;
    ev[idx] = v;
    atomicMax(&mkey[d * Hh + hh], f2key(v));   // global_atomic_max_i32
}

// ---- pass 2: ex = exp(e - m[dst]); segment sum over dst -------------------
__global__ __launch_bounds__(256) void k_edge_exp(const int* __restrict__ ei,
                                                  float* __restrict__ ev,
                                                  const int* __restrict__ mkey,
                                                  float* __restrict__ denom, int Hh) {
    int idx = blockIdx.x * blockDim.x + threadIdx.x;
    int tot = (EDGES + NODES) * Hh;
    if (idx >= tot) return;
    int e = idx / Hh, hh = idx - e * Hh;
    int s, d; edge_sd(ei, e, s, d); (void)s;
    float m = key2f(mkey[d * Hh + hh]);
    float x = __expf(ev[idx] - m);
    ev[idx] = x;
    atomicAdd(&denom[d * Hh + hh], x);         // global_atomic_add_f32
}

// ---- pass 3: out[dst] += (ex/denom[dst]) * h[src], vectorized over 4 ch ---
__global__ __launch_bounds__(256) void k_edge_aggr(const int* __restrict__ ei,
                                                   const float* __restrict__ ev,
                                                   const float* __restrict__ denom,
                                                   const float* __restrict__ hf,
                                                   float* __restrict__ outacc,
                                                   int Hh, int Cc) {
    int c4n = Cc >> 2;
    int idx = blockIdx.x * blockDim.x + threadIdx.x;
    int tot = (EDGES + NODES) * Hh * c4n;
    if (idx >= tot) return;
    int c4 = idx % c4n;
    int t  = idx / c4n;
    int hh = t % Hh;
    int e  = t / Hh;
    int s, d; edge_sd(ei, e, s, d);
    float alpha = ev[e * Hh + hh] / denom[d * Hh + hh];
    const float4 v = *((const float4*)(hf + ((size_t)s * Hh + hh) * Cc) + c4);
    float* op = outacc + ((size_t)d * Hh + hh) * Cc + (size_t)c4 * 4;
    atomicAdd(op + 0, alpha * v.x);
    atomicAdd(op + 1, alpha * v.y);
    atomicAdd(op + 2, alpha * v.z);
    atomicAdd(op + 3, alpha * v.w);
}

// ---- finalize: x = ELU(x + b) in place ------------------------------------
__global__ __launch_bounds__(256) void k_bias_elu(float* __restrict__ x,
                                                  const float* __restrict__ b,
                                                  int Nn, int F) {
    int idx = blockIdx.x * blockDim.x + threadIdx.x;
    if (idx >= Nn * F) return;
    float v = x[idx] + b[idx % F];
    x[idx] = v > 0.f ? v : expm1f(v);
}

// ---------------------------------------------------------------------------
extern "C" void kernel_launch(void* const* d_in, const int* in_sizes, int n_in,
                              void* d_out, int out_size, void* d_ws, size_t ws_size,
                              hipStream_t stream) {
    (void)in_sizes; (void)n_in; (void)out_size; (void)ws_size;
    const int N = NODES, E = EDGES;

    const float* x0 = (const float*)d_in[0];
    const int*   ei = (const int*)d_in[1];            // [2, E] row-major
    const float* W[5]; const float* AS[5]; const float* AD[5]; const float* B[5];
    for (int i = 0; i < 5; ++i) {
        W[i]  = (const float*)d_in[2 + 4 * i + 0];
        AS[i] = (const float*)d_in[2 + 4 * i + 1];
        AD[i] = (const float*)d_in[2 + 4 * i + 2];
        B[i]  = (const float*)d_in[2 + 4 * i + 3];
    }
    float* out = (float*)d_out;

    // d_out layout: [x_final N*64][rep0 N*1024][rep1][rep2][rep3][rep4 N*64]
    size_t offRep[5];
    offRep[0] = (size_t)N * 64;
    for (int i = 1; i <= 4; ++i) offRep[i] = offRep[i - 1] + (size_t)N * 1024;

    // ----- workspace partition (bytes, 256B aligned) -----
    char*  ws = (char*)d_ws;
    size_t o  = 0;
    auto carve = [&](size_t bytes) { void* p = ws + o; o += (bytes + 255) & ~(size_t)255; return p; };
    __bf16* xb    = (__bf16*)carve((size_t)2 * N * 1024);       // bf16 activations
    __bf16* wbT   = (__bf16*)carve((size_t)2 * 1024 * 1024);    // bf16 W^T [F][K]
    float*  hfeat = (float*) carve((size_t)4 * N * 1024);       // GEMM output h
    float*  als   = (float*) carve((size_t)4 * N * 8);
    float*  ald   = (float*) carve((size_t)4 * N * 8);
    int*    mkey  = (int*)   carve((size_t)4 * N * 8);
    float*  denom = (float*) carve((size_t)4 * N * 8);
    float*  evbuf = (float*) carve((size_t)4 * (E + N) * 8);

    const int Kd[5] = {256, 1024, 1024, 1024, 1024};
    const int Hd[5] = {8, 8, 8, 8, 1};
    const int Cd[5] = {128, 128, 128, 128, 64};

    for (int i = 0; i < 5; ++i) {
        const int K = Kd[i], Hh = Hd[i], Cc = Cd[i], F = Hh * Cc;
        const float* xin = (i == 0) ? x0 : (out + offRep[i - 1]);
        float* rep = out + offRep[i];            // accumulate + finalize here

        // 1) convert activations (row-major) and weights (transposed) to bf16
        size_t na = (size_t)N * K;
        k_f32_to_bf16<<<(unsigned)((na + 255) / 256), 256, 0, stream>>>(xin, xb, na);
        int nw = K * F;
        k_f32_to_bf16_T<<<(nw + 255) / 256, 256, 0, stream>>>(W[i], wbT, K, F);

        // 2) GEMM on the WMMA pipe: hfeat = xb @ W
        if (F == 1024) {
            dim3 gg(N / 32, F / 256);
            k_gemm_bf16_wmma<256, 4><<<gg, 256, 0, stream>>>(xb, wbT, hfeat, N, K, F);
        } else { // F == 64
            dim3 gg(N / 32, 1);
            k_gemm_bf16_wmma<64, 1><<<gg, 256, 0, stream>>>(xb, wbT, hfeat, N, K, F);
        }

        // 3) attention logits
        int nh = N * Hh;
        k_attn_logits<<<(nh + 255) / 256, 256, 0, stream>>>(hfeat, AS[i], AD[i], als, ald, N, Hh, Cc);

        // 4) init segment-max keys / denom / output accumulator
        k_init_attn<<<(nh + 255) / 256, 256, 0, stream>>>(mkey, denom, nh);
        size_t nf = (size_t)N * F;
        k_zero_f32<<<(unsigned)((nf + 255) / 256), 256, 0, stream>>>(rep, nf);

        // 5) edge passes: max -> exp/sum -> weighted aggregate
        int te = (E + N) * Hh;
        k_edge_max<<<(te + 255) / 256, 256, 0, stream>>>(ei, als, ald, evbuf, mkey, Hh);
        k_edge_exp<<<(te + 255) / 256, 256, 0, stream>>>(ei, evbuf, mkey, denom, Hh);
        int ta = (E + N) * Hh * (Cc >> 2);
        k_edge_aggr<<<(ta + 255) / 256, 256, 0, stream>>>(ei, evbuf, denom, hfeat, rep, Hh, Cc);

        // 6) bias + ELU in place (rep is this layer's output / next layer's input)
        k_bias_elu<<<(N * F + 255) / 256, 256, 0, stream>>>(rep, B[i], N, F);
    }

    // final output x == rep4
    size_t nfin = (size_t)N * 64;
    k_copy_f32<<<(unsigned)((nfin + 255) / 256), 256, 0, stream>>>(out, out + offRep[4], nfin);
}
